// MySimpleRNN_5171140625006
// MI455X (gfx1250) — compile-verified, matrix operand order checked by reading, not run on hardware
//
#include <hip/hip_runtime.h>
#include <hip/hip_bf16.h>

typedef __attribute__((ext_vector_type(16))) __bf16 v16bf;
typedef __attribute__((ext_vector_type(8)))  __bf16 v8bf;
typedef __attribute__((ext_vector_type(4)))  __bf16 v4bf;
typedef __attribute__((ext_vector_type(8)))  float  v8f;

// Explicit global (addrspace 1) pointer types: these survive asm laundering,
// guaranteeing global_load_b128 (LOADcnt only) instead of flat_load_b128
// (LOADcnt + DScnt joint tracking).
typedef __attribute__((address_space(1))) const __bf16 gbf;
typedef __attribute__((address_space(1))) const v8bf   gv8;

union V16 { v16bf v; v8bf h[2]; };

constexpr int NF      = 128;
constexpr int NH      = 512;
constexpr int TSTEPS  = 64;
constexpr int KTOT    = NH + NF;   // 640 : [h | x] concatenated K
constexpr int MT      = 32;        // batch rows per workgroup
constexpr int HSTRIDE = NH + 8;    // 520 (bank-conflict pad)
constexpr int XSTRIDE = NF + 8;    // 136
constexpr int HBUF    = MT * HSTRIDE;   // elements per h ping-pong buffer
constexpr int XBUF    = MT * XSTRIDE;   // elements per x ping-pong buffer

// LDS-side fragment load: two contiguous 16B chunks (ds_load_b128 x2).
__device__ __forceinline__ v16bf ldfragL(const __bf16* p) {
    V16 u;
    u.h[0] = *(const v8bf*)(p);
    u.h[1] = *(const v8bf*)(p + 16);
    return u.v;
}

// Global-side fragment load: two contiguous 16B chunks (global_load_b128 x2).
__device__ __forceinline__ v16bf ldfragG(gbf* p) {
    V16 u;
    u.h[0] = *(gv8*)(p);
    u.h[1] = *(gv8*)(p + 16);
    return u.v;
}

__device__ __forceinline__ v8f wmma_bf16(v16bf a, v16bf b, v8f c) {
    return __builtin_amdgcn_wmma_f32_16x16x32_bf16(
        /*neg_a=*/false, a, /*neg_b=*/false, b,
        /*c_mod=*/(short)0, c, /*reuse_a=*/false, /*reuse_b=*/false);
}

// tanh(v) = 1 - 2/(exp(2v)+1): 5 ops via TRANS v_exp_f32 + v_rcp_f32,
// branch-free, correct saturation at +/-inf. On the serial step critical
// path this replaces the ~25-op ocml tanhf expansion.
__device__ __forceinline__ float fast_tanh(float v) {
    float e = __expf(2.0f * v);
    return 1.0f - 2.0f * __builtin_amdgcn_rcpf(e + 1.0f);
}

// ---------------------------------------------------------------------------
// Prep: build W = [wh ; wx] transposed to [N=512][K=640] in bf16 so that
// B-fragments are two contiguous 16B chunks per lane (ISA 16-bit B layout).
// ---------------------------------------------------------------------------
__global__ __launch_bounds__(256)
void rnn_prep_weights(const float* __restrict__ wx,   // [NF][NH]
                      const float* __restrict__ wh,   // [NH][NH]
                      __bf16* __restrict__ W)         // [NH][KTOT]
{
    int idx = blockIdx.x * 256 + threadIdx.x;
    if (idx >= NH * KTOT) return;
    int n = idx / KTOT;
    int k = idx - n * KTOT;
    float v = (k < NH) ? wh[(size_t)k * NH + n] : wx[(size_t)(k - NH) * NH + n];
    W[idx] = (__bf16)v;
}

// ---------------------------------------------------------------------------
// Fused recurrence: each workgroup owns MT batch rows for all T steps.
//   h' = tanh([h | x_t] @ W + b),   W pre-transposed/concatenated in L2.
// LDS ping-pong (read buf t&1, write buf 1-(t&1)) -> single barrier/step.
// ---------------------------------------------------------------------------
__global__ __launch_bounds__(256)
void rnn_scan_kernel(const float*  __restrict__ x,    // [B][T][NF] f32
                     const __bf16* __restrict__ W,    // [NH][KTOT] bf16
                     const float*  __restrict__ bias, // [NH]
                     float*        __restrict__ out)  // [B][NH]
{
    __shared__ __align__(16) __bf16 h_lds[2 * HBUF];
    __shared__ __align__(16) __bf16 x_lds[2 * XBUF];

    const int tid  = threadIdx.x;
    const int wave = tid >> 5;
    const int lane = tid & 31;
    const int l16  = lane & 15;
    const int hi   = (lane >> 4) & 1;       // half-wave: K sub-block select
    const int row0 = blockIdx.x * MT;       // first batch row of this WG

    // h0 = 0 (buffer 0 only)
    for (int i = tid; i < HBUF; i += 256) h_lds[i] = (__bf16)0.0f;

    // stage x_t for t=0 into x buffer 0: 32 rows x 128 f32, 8 threads/row
    const int sr = tid >> 3;    // staging row 0..31
    const int sc = tid & 7;     // staging column group
    const float* xstage = x + ((size_t)(row0 + sr) * TSTEPS) * NF;  // t=0
    {
        #pragma unroll
        for (int i = 0; i < 4; ++i) {
            int f4 = sc + i * 8;
            float4 v = ((const float4*)xstage)[f4];
            v4bf bv4 = { (__bf16)v.x, (__bf16)v.y, (__bf16)v.z, (__bf16)v.w };
            *(v4bf*)&x_lds[sr * XSTRIDE + f4 * 4] = bv4;
        }
    }
    __syncthreads();

    // Wave w owns N-tiles [4w .. 4w+3] and both M-tiles (rows 0-15, 16-31).
    const int nt0 = wave * 4;
    float bv[4];
    #pragma unroll
    for (int j = 0; j < 4; ++j) bv[j] = bias[(nt0 + j) * 16 + l16];

    // Invariant lane-adjusted fragment base pointers (buffer 0).
    const __bf16* hA0 = h_lds + (size_t)(0  + l16) * HSTRIDE + hi * 8;
    const __bf16* hA1 = h_lds + (size_t)(16 + l16) * HSTRIDE + hi * 8;
    const __bf16* xA0 = x_lds + (size_t)(0  + l16) * XSTRIDE + hi * 8;
    const __bf16* xA1 = x_lds + (size_t)(16 + l16) * XSTRIDE + hi * 8;

    // Per-lane element offsets into W for the 4 owned N-tile columns
    // (32-bit: pairs with SGPR base for GVS global addressing).
    unsigned boff[4];
    #pragma unroll
    for (int j = 0; j < 4; ++j)
        boff[j] = (unsigned)(((nt0 + j) * 16 + l16) * KTOT + hi * 8);

    // Global (addrspace 1) weight base. Laundered through an SGPR asm each t
    // iteration so LICM cannot hoist 640 VGPRs of weights out of the time
    // loop (that caused scratch spills); the 640KB W stays L2-resident, so
    // re-issuing global_load_b128 per step is the right tradeoff.
    gbf* Wg = (gbf*)(uintptr_t)W;

    v8f acc[2][4];

    for (int t = 0; t < TSTEPS; ++t) {
        asm volatile("" : "+s"(Wg));   // defeat LICM on weight loads

        const int pb     = t & 1;                 // read buffer select
        const int roff_h = pb ? HBUF : 0;
        const int woff_h = HBUF - roff_h;
        const int roff_x = pb ? XBUF : 0;
        const int woff_x = XBUF - roff_x;

        gbf* Bq0 = Wg + boff[0];
        gbf* Bq1 = Wg + boff[1];
        gbf* Bq2 = Wg + boff[2];
        gbf* Bq3 = Wg + boff[3];

        // init accumulators with bias (16x16 f32 C layout: a lane's 8 VGPRs
        // all share N = tile*16 + l16)
        #pragma unroll
        for (int j = 0; j < 4; ++j) {
            v8f b8 = { bv[j], bv[j], bv[j], bv[j], bv[j], bv[j], bv[j], bv[j] };
            acc[0][j] = b8;
            acc[1][j] = b8;
        }

        // warm L2 for next step's x tile
        if (t + 1 < TSTEPS) {
            __builtin_prefetch(xstage + NF + sc * 16, 0, 1);
        }

        // K loop: 16 iters over h (K=512) + 4 iters over x_t (K=128)
        #pragma unroll
        for (int kk = 0; kk < KTOT / 32; ++kk) {
            const int k0 = kk * 32;
            v16bf a0, a1;
            if (kk < NH / 32) {
                a0 = ldfragL(hA0 + roff_h + k0);
                a1 = ldfragL(hA1 + roff_h + k0);
            } else {
                a0 = ldfragL(xA0 + roff_x + (k0 - NH));
                a1 = ldfragL(xA1 + roff_x + (k0 - NH));
            }
            v16bf b0 = ldfragG(Bq0 + k0);
            acc[0][0] = wmma_bf16(a0, b0, acc[0][0]);
            acc[1][0] = wmma_bf16(a1, b0, acc[1][0]);
            v16bf b1 = ldfragG(Bq1 + k0);
            acc[0][1] = wmma_bf16(a0, b1, acc[0][1]);
            acc[1][1] = wmma_bf16(a1, b1, acc[1][1]);
            v16bf b2 = ldfragG(Bq2 + k0);
            acc[0][2] = wmma_bf16(a0, b2, acc[0][2]);
            acc[1][2] = wmma_bf16(a1, b2, acc[1][2]);
            v16bf b3 = ldfragG(Bq3 + k0);
            acc[0][3] = wmma_bf16(a0, b3, acc[0][3]);
            acc[1][3] = wmma_bf16(a1, b3, acc[1][3]);
        }

        // activation in place (keeps register footprint flat)
        #pragma unroll
        for (int mt = 0; mt < 2; ++mt)
            #pragma unroll
            for (int j = 0; j < 4; ++j)
                #pragma unroll
                for (int r = 0; r < 8; ++r)
                    acc[mt][j][r] = fast_tanh(acc[mt][j][r]);

        // h' -> other LDS buffer as bf16; no barrier needed before writing
        // (readers use the ping buffer, writers the pong buffer).
        #pragma unroll
        for (int mt = 0; mt < 2; ++mt)
            #pragma unroll
            for (int j = 0; j < 4; ++j) {
                const int n = (nt0 + j) * 16 + l16;
                #pragma unroll
                for (int r = 0; r < 8; ++r) {
                    const int m = mt * 16 + hi * 8 + r;
                    h_lds[woff_h + m * HSTRIDE + n] = (__bf16)acc[mt][j][r];
                }
            }

        // stage x_{t+1} into the other x buffer
        if (t + 1 < TSTEPS) {
            xstage += NF;
            #pragma unroll
            for (int i = 0; i < 4; ++i) {
                int f4 = sc + i * 8;
                float4 v = ((const float4*)xstage)[f4];
                v4bf bv4 = { (__bf16)v.x, (__bf16)v.y, (__bf16)v.z, (__bf16)v.w };
                *(v4bf*)&x_lds[woff_x + sr * XSTRIDE + f4 * 4] = bv4;
            }
        }

        __syncthreads();   // single barrier per step: writes visible for t+1
    }

    // final h (t = T-1) is in acc (f32, post-tanh) -> d_out
    #pragma unroll
    for (int mt = 0; mt < 2; ++mt)
        #pragma unroll
        for (int j = 0; j < 4; ++j) {
            const int n = (nt0 + j) * 16 + l16;
            #pragma unroll
            for (int r = 0; r < 8; ++r) {
                const size_t m = (size_t)row0 + mt * 16 + hi * 8 + r;
                out[m * NH + n] = acc[mt][j][r];
            }
        }
}

extern "C" void kernel_launch(void* const* d_in, const int* in_sizes, int n_in,
                              void* d_out, int out_size, void* d_ws, size_t ws_size,
                              hipStream_t stream) {
    const float* x    = (const float*)d_in[0];   // [B][T][NF]
    const float* wx   = (const float*)d_in[1];   // [NF][NH]
    const float* wh   = (const float*)d_in[2];   // [NH][NH]
    const float* bias = (const float*)d_in[3];   // [1][NH]
    float* out        = (float*)d_out;           // [B][NH]

    const int Bbatch = in_sizes[0] / (TSTEPS * NF);   // 4096

    __bf16* W = (__bf16*)d_ws;                   // [NH][KTOT] bf16 = 640 KB

    {
        int total  = NH * KTOT;
        int blocks = (total + 255) / 256;
        rnn_prep_weights<<<blocks, 256, 0, stream>>>(wx, wh, W);
    }
    {
        int blocks = Bbatch / MT;                // 128 workgroups
        rnn_scan_kernel<<<blocks, 256, 0, stream>>>(x, W, bias, out);
    }
}
